// CrossAttention_12773232739141
// MI455X (gfx1250) — matrix operand order repaired and checked by reading, not compiled, not run on hardware
//
#include <hip/hip_runtime.h>

// ---------------------------------------------------------------------------
// CDNA5 (gfx1250) cross-attention pipeline.
// bf16 WMMA (f32 accum) + TDM tensor_load_to_lds GEMM staging +
// async global->LDS attention staging + DPP16 softmax reductions.
// ---------------------------------------------------------------------------

typedef __bf16 bf16;
typedef __attribute__((ext_vector_type(4)))  __bf16 bf16x4;
typedef __attribute__((ext_vector_type(8)))  __bf16 bf16x8;
typedef __attribute__((ext_vector_type(16))) __bf16 bf16x16;
typedef __attribute__((ext_vector_type(8)))  float  f32x8;
typedef __attribute__((ext_vector_type(4)))  unsigned int u32x4;
typedef __attribute__((ext_vector_type(8)))  unsigned int u32x8;
typedef __attribute__((ext_vector_type(4)))  int i32x4;
typedef __attribute__((ext_vector_type(8)))  int i32x8;

__device__ __forceinline__ f32x8 wmma_bf16(bf16x16 a, bf16x16 b, f32x8 c) {
  // 8-arg form: (neg_a, A, neg_b, B, c_mod, C, reuse_a, reuse_b)
  return __builtin_amdgcn_wmma_f32_16x16x32_bf16(false, a, false, b, (short)0, c,
                                                 false, false);
}

__device__ __forceinline__ bf16x16 cat8(bf16x8 lo, bf16x8 hi) {
  return __builtin_shufflevector(lo, hi, 0, 1, 2, 3, 4, 5, 6, 7,
                                 8, 9, 10, 11, 12, 13, 14, 15);
}

__device__ __forceinline__ f32x8 zero8() {
  f32x8 z = {0.f, 0.f, 0.f, 0.f, 0.f, 0.f, 0.f, 0.f};
  return z;
}

// A-matrix fragment (16x32 bf16, ISA 7.12.2): lane m (m=lane&15) holds row m.
// lanes 0-15: K = {ko..ko+7, ko+16..ko+23} with ko=0; lanes 16-31: ko=8.
__device__ __forceinline__ bf16x16 ld_frag_a(const bf16* p, int ko) {
  bf16x8 lo = *(const bf16x8*)(p + ko);
  bf16x8 hi = *(const bf16x8*)(p + ko + 16);
  return cat8(lo, hi);
}

// B-matrix fragment (32x16 bf16): lane n (n=lane&15) holds column n;
// lanes 0-15: K = {ko..ko+7, ko+16..ko+23} with ko=0; lanes 16-31: ko=8.
__device__ __forceinline__ bf16x16 ld_frag_b(const bf16* p, int ko) {
  bf16x8 lo = *(const bf16x8*)(p + ko);
  bf16x8 hi = *(const bf16x8*)(p + ko + 16);
  return cat8(lo, hi);
}

// CDNA5 async DMA: memory -> LDS, no VGPR data, tracked by ASYNCcnt.
__device__ __forceinline__ void async_b128(const bf16* g, const bf16* l) {
  asm volatile("global_load_async_to_lds_b128 %0, %1, off"
               :: "v"((unsigned int)(size_t)l), "v"(g)
               : "memory");
}

// ---------------------------------------------------------------------------
// Tensor Data Mover: 2D tile (tile_d0 x tile_d1 bf16) from a row-major tensor
// (row stride stride0 elems) into LDS at lds_addr.  LDS rows are padded by
// 8 bf16 (pad_interval=3 -> every 16 DWORDs = one 64B row; pad_amount=3 ->
// 4 DWORDs = 16B), reproducing the 40-element row stride the WMMA fragment
// readers use.  D# bit layout per CDNA5 ISA.
// ---------------------------------------------------------------------------
__device__ __forceinline__ void tdm_load_2d(const bf16* gaddr, unsigned lds_addr,
                                            unsigned tensor_d0, unsigned tensor_d1,
                                            unsigned tile_d0, unsigned tile_d1,
                                            unsigned stride0) {
  const unsigned long long ga = (unsigned long long)(size_t)gaddr;
  u32x4 g0;
  g0[0] = 1u;                                     // count=1 (valid), user mode
  g0[1] = lds_addr;                               // lds_addr [63:32]
  g0[2] = (unsigned)(ga & 0xFFFFFFFFu);           // global_addr [95:64]
  g0[3] = (unsigned)((ga >> 32) & 0x01FFFFFFu)    // global_addr [120:96]
          | (2u << 30);                           // type=2 ("image") [127:126]
  u32x8 g1;
  g1[0] = (1u << 16)                              // data_size = 2 bytes
          | (1u << 20)                            // pad_enable
          | (3u << 22)                            // pad_interval: 16 DWORDs
          | (3u << 25);                           // pad_amount:   4 DWORDs
  g1[1] = (tensor_d0 & 0xFFFFu) << 16;            // tensor_dim0[15:0] @ 63:48
  g1[2] = ((tensor_d0 >> 16) & 0xFFFFu)           // tensor_dim0[31:16]
          | ((tensor_d1 & 0xFFFFu) << 16);        // tensor_dim1[15:0]
  g1[3] = ((tensor_d1 >> 16) & 0xFFFFu)
          | (tile_d0 << 16);                      // tile_dim0 @ 127:112
  g1[4] = tile_d1;                                // tile_dim1 (tile_dim2 = 0)
  g1[5] = stride0;                                // tensor_dim0_stride[31:0]
  g1[6] = 0;                                      // stride[47:32] | dim1_stride lo
  g1[7] = 0;
  const i32x4 z4 = {0, 0, 0, 0};
  const i32x8 g1s = __builtin_bit_cast(i32x8, g1);
#if __clang_major__ >= 23
  const i32x8 z8 = {0, 0, 0, 0, 0, 0, 0, 0};
  __builtin_amdgcn_tensor_load_to_lds(g0, g1s, z4, z4, z8, 0);
#else
  __builtin_amdgcn_tensor_load_to_lds(g0, g1s, z4, z4, 0);
#endif
}

// 16-lane butterfly reductions via DPP16 (no LDS, no waits).
template <int CTRL>
__device__ __forceinline__ float dppf(float v) {
  int s = __builtin_bit_cast(int, v);
  return __builtin_bit_cast(
      float, __builtin_amdgcn_update_dpp(s, s, CTRL, 0xF, 0xF, false));
}
__device__ __forceinline__ float redmax16(float x) {
  x = fmaxf(x, dppf<0xB1>(x));   // quad_perm(1,0,3,2)
  x = fmaxf(x, dppf<0x4E>(x));   // quad_perm(2,3,0,1)
  x = fmaxf(x, dppf<0x141>(x));  // row_half_mirror
  x = fmaxf(x, dppf<0x140>(x));  // row_mirror
  return x;
}
__device__ __forceinline__ float redsum16(float x) {
  x += dppf<0xB1>(x);
  x += dppf<0x4E>(x);
  x += dppf<0x141>(x);
  x += dppf<0x140>(x);
  return x;
}

// ---------------------------------------------------------------------------
// f32 -> bf16 conversion (vectorized x4)
// ---------------------------------------------------------------------------
__global__ void cvt_f32_to_bf16(const float* __restrict__ in,
                                bf16* __restrict__ out, int n4) {
  int i = blockIdx.x * blockDim.x + threadIdx.x;
  if (i >= n4) return;
  float4 f = ((const float4*)in)[i];
  bf16x4 o;
  o.x = (bf16)f.x; o.y = (bf16)f.y; o.z = (bf16)f.z; o.w = (bf16)f.w;
  ((bf16x4*)out)[i] = o;
}

// ---------------------------------------------------------------------------
// GEMM: C[M,N] = A[M,K] * W[N,K]^T   (A, W bf16 row-major, K contiguous)
// 128x128 block tile, 8 waves (4x2), wave tile 32x64 = 2x4 WMMA accumulators.
// K-panels staged by the Tensor Data Mover: wave 0 issues one descriptor per
// panel (TDM ignores EXEC; one instruction moves the whole 8KB strided tile),
// waits on TENSORcnt, and the workgroup barrier publishes the LDS data to the
// other waves.  Double-buffered: TENSORcnt<=2 means "panel t landed, panel
// t+1 still in flight" (TDM ops complete in order per wave).
// ---------------------------------------------------------------------------
template <bool F32OUT>
__global__ __launch_bounds__(256) void gemm_xt(const bf16* __restrict__ A,
                                               const bf16* __restrict__ W,
                                               bf16* __restrict__ Cb,
                                               float* __restrict__ Cf,
                                               const float* __restrict__ bias,
                                               int M, int N, int K) {
  constexpr int BM = 128, BN = 128, BK = 32, LSTR = BK + 8;  // 80B rows, 16B aligned
  __shared__ bf16 As[2][BM][LSTR];
  __shared__ bf16 Bs[2][BN][LSTR];

  const int tid  = threadIdx.x;
  const int lane = tid & 31;
  const int wv   = tid >> 5;
  const int wm   = wv >> 1;      // 0..3  (M direction)
  const int wn   = wv & 1;       // 0..1  (N direction)
  const int lm   = lane & 15;
  const int lh   = lane >> 4;
  const int mBase = blockIdx.y * BM;
  const int nBase = blockIdx.x * BN;

  auto issue = [&](int k0, int buf) {
    if (wv == 0) {
      tdm_load_2d(A + (size_t)mBase * K + k0, (unsigned)(size_t)&As[buf][0][0],
                  (unsigned)K, BM, BK, BM, (unsigned)K);
      tdm_load_2d(W + (size_t)nBase * K + k0, (unsigned)(size_t)&Bs[buf][0][0],
                  (unsigned)K, BN, BK, BN, (unsigned)K);
    }
  };

  f32x8 acc[2][4];
#pragma unroll
  for (int i = 0; i < 2; ++i)
#pragma unroll
    for (int j = 0; j < 4; ++j) acc[i][j] = zero8();

  const int T = K / BK;
  issue(0, 0);
  if (T > 1) issue(BK, 1);

  for (int t = 0; t < T; ++t) {
    if (wv == 0) {
      if (t + 1 < T) __builtin_amdgcn_s_wait_tensorcnt((short)2);
      else           __builtin_amdgcn_s_wait_tensorcnt((short)0);
    }
    __syncthreads();  // wave0's tensorcnt wait + barrier -> panel t visible to all

    const bf16(*as)[LSTR] = As[t & 1];
    const bf16(*bs)[LSTR] = Bs[t & 1];
    bf16x16 af[2], bfr[4];
#pragma unroll
    for (int i = 0; i < 2; ++i)
      af[i] = ld_frag_a(&as[wm * 32 + i * 16 + lm][0], lh * 8);
#pragma unroll
    for (int j = 0; j < 4; ++j)
      bfr[j] = ld_frag_b(&bs[wn * 64 + j * 16 + lm][0], lh * 8);
#pragma unroll
    for (int i = 0; i < 2; ++i)
#pragma unroll
      for (int j = 0; j < 4; ++j) acc[i][j] = wmma_bf16(af[i], bfr[j], acc[i][j]);

    __syncthreads();  // buffer t&1 fully consumed by all waves
    if (t + 2 < T) issue((t + 2) * BK, t & 1);
  }

  // C/D layout (ISA 7.12.2): VGPR r, lanes0-15 -> M=r,N=lane; lanes16-31 -> M=8+r.
#pragma unroll
  for (int i = 0; i < 2; ++i) {
#pragma unroll
    for (int j = 0; j < 4; ++j) {
      const int mg = mBase + wm * 32 + i * 16 + lh * 8;
      const int ng = nBase + wn * 64 + j * 16 + lm;
      if constexpr (F32OUT) {
        const float bb = bias[ng];
#pragma unroll
        for (int r = 0; r < 8; ++r)
          Cf[(size_t)(mg + r) * N + ng] = acc[i][j][r] + bb;
      } else {
#pragma unroll
        for (int r = 0; r < 8; ++r)
          Cb[(size_t)(mg + r) * N + ng] = (bf16)acc[i][j][r];
      }
    }
  }
}

// ---------------------------------------------------------------------------
// Flash attention: block = (b, h, 128 q rows), 8 waves x 16 q-rows each.
// K tile staged via async DMA; V transposed through VGPRs into Vt[d][k] so
// both WMMA B-fragments are contiguous ds_load_b128s. Online softmax in f32
// with DPP16 butterfly reductions and raw v_exp_f32.
// ---------------------------------------------------------------------------
__global__ __launch_bounds__(256) void attn_kernel(const bf16* __restrict__ Q,
                                                   const bf16* __restrict__ KV,
                                                   bf16* __restrict__ O,
                                                   int ktiles) {
  constexpr int VSTR = 72;  // 144B row stride, 16B aligned
  __shared__ bf16 Ks[64][VSTR];       // [key][d]
  __shared__ bf16 Vt[64][VSTR];       // [d][key]  (transposed V)
  __shared__ bf16 Ps[8][16][VSTR];    // per-wave P tile, [m][key]

  const int tid  = threadIdx.x;
  const int lane = tid & 31;
  const int wv   = tid >> 5;
  const int lm   = lane & 15;
  const int lh   = lane >> 4;
  const int qblk = blockIdx.x, h = blockIdx.y, b = blockIdx.z;

  const bf16* qrow  = Q + (size_t)(b * 2048 + qblk * 128 + wv * 16 + lm) * 1024 + h * 64;
  const bf16* kbase = KV + (size_t)b * 2048 * 2048 + h * 64;
  const bf16* vbase = kbase + 1024;

  // Q fragments for this wave's 16 rows (d = 0..31 and 32..63), live in VGPRs.
  bf16x16 qf[2];
  qf[0] = ld_frag_a(qrow, lh * 8);
  qf[1] = ld_frag_a(qrow + 32, lh * 8);

  f32x8 Oacc[4];
#pragma unroll
  for (int dt = 0; dt < 4; ++dt) Oacc[dt] = zero8();
  float m_i[8], l_i[8];
#pragma unroll
  for (int r = 0; r < 8; ++r) { m_i[r] = -3.0e38f; l_i[r] = 0.f; }

  constexpr float SC2 = 0.125f * 1.44269504088896f;  // SCALE * log2(e)

  for (int kt = 0; kt < ktiles; ++kt) {
    __syncthreads();  // tiles from previous iteration fully consumed
#pragma unroll
    for (int c = 0; c < 2; ++c) {
      int chunk = tid + c * 256;         // 0..511
      int kr = chunk >> 3;               // key row 0..63
      int dc = (chunk & 7) * 8;          // d chunk
      size_t goff = (size_t)(kt * 64 + kr) * 2048 + dc;
      async_b128(kbase + goff, &Ks[kr][dc]);       // K: DMA straight to LDS
      bf16x8 vv = *(const bf16x8*)(vbase + goff);  // V: through VGPRs (transpose)
#pragma unroll
      for (int u = 0; u < 8; ++u) Vt[dc + u][kr] = vv[u];
    }
    asm volatile("s_wait_asynccnt 0x0" ::: "memory");
    __syncthreads();  // fence also waits V's loadcnt/dscnt

    // S(16x64) = Q(16x64) @ K_tile^T : 4 column tiles, 2 k-steps each.
    f32x8 S[4];
#pragma unroll
    for (int j = 0; j < 4; ++j) {
      f32x8 s = zero8();
#pragma unroll
      for (int ks = 0; ks < 2; ++ks) {
        bf16x16 bk = ld_frag_b(&Ks[j * 16 + lm][0], ks * 32 + lh * 8);
        s = wmma_bf16(qf[ks], bk, s);
      }
      S[j] = s;
    }

    // Online softmax.  Row r lives in VGPR r across a 16-lane half
    // (lanes 0-15 -> row r, lanes 16-31 -> row r+8); DPP16 stays in-half.
#pragma unroll
    for (int r = 0; r < 8; ++r) {
      float mx = fmaxf(fmaxf(S[0][r], S[1][r]), fmaxf(S[2][r], S[3][r]));
      mx = redmax16(mx);
      const float nm   = fmaxf(m_i[r], mx);
      const float corr = __builtin_amdgcn_exp2f((m_i[r] - nm) * SC2);
      m_i[r] = nm;
      float rs = 0.f;
#pragma unroll
      for (int j = 0; j < 4; ++j) {
        float p = __builtin_amdgcn_exp2f((S[j][r] - nm) * SC2);
        S[j][r] = p;
        rs += p;
      }
      rs = redsum16(rs);
      l_i[r] = l_i[r] * corr + rs;
#pragma unroll
      for (int dt = 0; dt < 4; ++dt) Oacc[dt][r] *= corr;
    }

    // P (C-layout f32) -> per-wave LDS tile in A-matrix-friendly row-major bf16.
#pragma unroll
    for (int j = 0; j < 4; ++j)
#pragma unroll
      for (int r = 0; r < 8; ++r)
        Ps[wv][lh * 8 + r][j * 16 + lm] = (bf16)S[j][r];
    asm volatile("s_wait_dscnt 0x0" ::: "memory");  // RAW through per-wave LDS

    // O(16x64) += P(16x64) @ V_tile(64x64): 4 d tiles x 2 k-steps.
#pragma unroll
    for (int dt = 0; dt < 4; ++dt) {
      f32x8 o = Oacc[dt];
#pragma unroll
      for (int ks = 0; ks < 2; ++ks) {
        bf16x16 a  = ld_frag_a(&Ps[wv][lm][0], ks * 32 + lh * 8);
        bf16x16 bv = ld_frag_b(&Vt[dt * 16 + lm][0], ks * 32 + lh * 8);
        o = wmma_bf16(a, bv, o);
      }
      Oacc[dt] = o;
    }
  }

  // Normalize and store O in bf16.
  bf16* orow = O + (size_t)(b * 2048 + qblk * 128 + wv * 16 + lh * 8) * 1024 +
               h * 64 + lm;
#pragma unroll
  for (int dt = 0; dt < 4; ++dt)
#pragma unroll
    for (int r = 0; r < 8; ++r)
      orow[(size_t)r * 1024 + dt * 16] = (bf16)(Oacc[dt][r] / l_i[r]);
}

// ---------------------------------------------------------------------------
// Launch
// ---------------------------------------------------------------------------
extern "C" void kernel_launch(void* const* d_in, const int* in_sizes, int n_in,
                              void* d_out, int out_size, void* d_ws, size_t ws_size,
                              hipStream_t stream) {
  (void)in_sizes; (void)n_in; (void)out_size; (void)ws_size;
  const float* x     = (const float*)d_in[0];
  const float* ctx   = (const float*)d_in[1];
  const float* Wq    = (const float*)d_in[2];
  const float* Wkv   = (const float*)d_in[3];
  const float* Wproj = (const float*)d_in[4];
  const float* bproj = (const float*)d_in[5];
  float* out = (float*)d_out;

  const size_t NTOK = 4 * 2048;        // 8192 tokens (x and context)
  const size_t NX   = NTOK * 1024;     // x / q / o element counts
  const size_t NKV  = NTOK * 2048;
  const size_t NWQ  = 1024 * 1024, NWKV = 2048 * 1024, NWP = 1024 * 1024;

  char* ws = (char*)d_ws;
  size_t off = 0;
  auto carve = [&](size_t elems) -> bf16* {
    bf16* p = (bf16*)(ws + off);
    off += (elems * sizeof(bf16) + 255) & ~(size_t)255;
    return p;
  };
  bf16* xb   = carve(NX);
  bf16* cb   = carve(NX);
  bf16* wqb  = carve(NWQ);
  bf16* wkvb = carve(NWKV);
  bf16* wpb  = carve(NWP);
  bf16* qb   = carve(NX);
  bf16* kvb  = carve(NKV);
  bf16* ob   = carve(NX);

  auto cvt = [&](const float* src, bf16* dst, size_t n) {
    int n4 = (int)(n / 4);
    cvt_f32_to_bf16<<<dim3((n4 + 255) / 256), dim3(256), 0, stream>>>(src, dst, n4);
  };
  cvt(x,     xb,   NX);
  cvt(ctx,   cb,   NX);
  cvt(Wq,    wqb,  NWQ);
  cvt(Wkv,   wkvb, NWKV);
  cvt(Wproj, wpb,  NWP);

  // Q = x @ Wq^T            (8192 x 1024) bf16
  gemm_xt<false><<<dim3(1024 / 128, 8192 / 128), dim3(256), 0, stream>>>(
      xb, wqb, qb, nullptr, nullptr, 8192, 1024, 1024);
  // KV = context @ Wkv^T    (8192 x 2048) bf16
  gemm_xt<false><<<dim3(2048 / 128, 8192 / 128), dim3(256), 0, stream>>>(
      cb, wkvb, kvb, nullptr, nullptr, 8192, 2048, 1024);
  // Flash attention -> O bf16
  attn_kernel<<<dim3(2048 / 128, 16, 4), dim3(256), 0, stream>>>(qb, kvb, ob,
                                                                 2048 / 64);
  // out = O @ Wproj^T + bproj  (f32)
  gemm_xt<true><<<dim3(1024 / 128, 8192 / 128), dim3(256), 0, stream>>>(
      ob, wpb, nullptr, out, bproj, 8192, 1024, 1024);
}
